// HieraMultiScaleAttention_20040317403205
// MI455X (gfx1250) — matrix operand order repaired and checked by reading, not compiled, not run on hardware
//
#include <hip/hip_runtime.h>

typedef __attribute__((ext_vector_type(16))) _Float16 v16h;
typedef __attribute__((ext_vector_type(8)))  _Float16 v8h;
typedef __attribute__((ext_vector_type(8)))  float    v8f;

#define B_    8
#define W_    56
#define DIM_  96
#define DOUT_ 192
#define NH_   3
#define HD_   64
#define HW_   3136   // 56*56
#define QN_   784    // 28*28
#define NQKV_ 576    // 3*192

__device__ __forceinline__ v8f wmma16(v16h a, v16h b, v8f c) {
  return __builtin_amdgcn_wmma_f32_16x16x32_f16(false, a, false, b, (short)0, c, false, false);
}

__device__ __forceinline__ float redmax16(float v) {
  v = fmaxf(v, __shfl_xor(v, 1, 32));
  v = fmaxf(v, __shfl_xor(v, 2, 32));
  v = fmaxf(v, __shfl_xor(v, 4, 32));
  v = fmaxf(v, __shfl_xor(v, 8, 32));
  return v;
}
__device__ __forceinline__ float redsum16(float v) {
  v += __shfl_xor(v, 1, 32);
  v += __shfl_xor(v, 2, 32);
  v += __shfl_xor(v, 4, 32);
  v += __shfl_xor(v, 8, 32);
  return v;
}

// Issue one per-lane 16B async copy global -> LDS (tracked by ASYNCcnt).
__device__ __forceinline__ void async_b128(unsigned lds_off, const void* gaddr) {
  asm volatile("global_load_async_to_lds_b128 %0, %1, off"
               :: "v"(lds_off), "v"(gaddr) : "memory");
}

// ---------------------------------------------------------------------------
// Kernel 1: qkv = x @ W_qkv + b_qkv.  Grid (nt=36, ptgrp=49, b=8), 4 waves
// per block, one 16x16 output tile per wave, K=96 -> 3 WMMA.
// q/k written head-major (bh,pos,d); V written TRANSPOSED (bh,d,pos).
// ---------------------------------------------------------------------------
__global__ __launch_bounds__(128) void qkv_gemm(
    const float* __restrict__ x, const float* __restrict__ Wq,
    const float* __restrict__ bq,
    _Float16* __restrict__ qfull, _Float16* __restrict__ kf,
    _Float16* __restrict__ vt)
{
  const int lane  = threadIdx.x & 31;
  const int col16 = lane & 15;
  const int g     = lane >> 4;
  const int nt    = blockIdx.x;                          // 0..35
  const int pt    = blockIdx.y * 4 + (threadIdx.x >> 5); // 0..195
  const int b     = blockIdx.z;                          // 0..7
  const int arow  = b * HW_ + pt * 16 + col16;
  const int ncol  = nt * 16 + col16;

  v8f acc = {};
  #pragma unroll
  for (int kc = 0; kc < 3; ++kc) {
    const int kb = kc * 32;
    v16h a, bb;
    const float* ap = x + (size_t)arow * DIM_ + kb + g * 8;
    #pragma unroll
    for (int e = 0; e < 8; ++e) { a[e] = (_Float16)ap[e]; a[8 + e] = (_Float16)ap[16 + e]; }
    #pragma unroll
    for (int e = 0; e < 16; ++e)
      bb[e] = (_Float16)Wq[(size_t)(kb + g * 8 + e + (e & 8)) * NQKV_ + ncol];
    acc = wmma16(a, bb, acc);
  }

  const float bias = bq[ncol];
  const int which = (nt * 16) / DOUT_;            // 0=q 1=k 2=v (uniform)
  const int head  = ((nt * 16) % DOUT_) / HD_;    // uniform
  const int d     = (nt % 4) * 16 + col16;
  const int bh    = b * NH_ + head;
  if (which == 2) {
    v8h pk;
    #pragma unroll
    for (int r = 0; r < 8; ++r) pk[r] = (_Float16)(acc[r] + bias);
    *(v8h*)(vt + ((size_t)bh * HD_ + d) * HW_ + pt * 16 + 8 * g) = pk;
  } else {
    _Float16* dst = (which == 0) ? qfull : kf;
    #pragma unroll
    for (int r = 0; r < 8; ++r) {
      const int pos = pt * 16 + r + 8 * g;        // C/D: row = r + 8*group
      dst[((size_t)bh * HW_ + pos) * HD_ + d] = (_Float16)(acc[r] + bias);
    }
  }
}

// ---------------------------------------------------------------------------
// Kernel 2: 2x2 max-pool of q + 1/sqrt(64) scale.
// ---------------------------------------------------------------------------
__global__ __launch_bounds__(256) void pool_q(
    const _Float16* __restrict__ qfull, _Float16* __restrict__ qp)
{
  const int idx = blockIdx.x * 256 + threadIdx.x;   // 24*784*64 exactly
  const int d   = idx & 63;
  const int p   = (idx >> 6) % QN_;
  const int bh  = idx / (QN_ * 64);
  const int hq  = p / 28, wq = p % 28;
  const size_t base = (size_t)bh * HW_ * HD_;
  float m = -1e30f;
  #pragma unroll
  for (int dy = 0; dy < 2; ++dy)
    #pragma unroll
    for (int dx = 0; dx < 2; ++dx) {
      const int pos = (2 * hq + dy) * W_ + (2 * wq + dx);
      m = fmaxf(m, (float)qfull[base + (size_t)pos * HD_ + d]);
    }
  qp[idx] = (_Float16)(m * 0.125f);
}

// ---------------------------------------------------------------------------
// Kernel 3: fused flash attention.  Grid (7, 24): 7 waves/block all share one
// (b,head); each wave owns one 16-row q tile.  Per 32-key step the block
// double-buffers the 4KB K tile + 4KB V tile into LDS with
// global_load_async_to_lds_b128 (ASYNCcnt), then runs 8 WMMA + online
// softmax out of LDS while the next tiles stream in.
// ---------------------------------------------------------------------------
__global__ __launch_bounds__(224) void flash_attn(
    const _Float16* __restrict__ qp, const _Float16* __restrict__ kf,
    const _Float16* __restrict__ vt, _Float16* __restrict__ attn_out)
{
  __shared__ _Float16 Kbuf[2][32 * 64];  // 2 x 4KB, rows = 32 keys of 64 halves
  __shared__ _Float16 Vbuf[2][64 * 32];  // 2 x 4KB, rows = 64 dims of 32 pos
  __shared__ _Float16 pld[7][16 * 32];   // wave-private P staging
  const int tid   = threadIdx.x;
  const int lane  = tid & 31;
  const int col16 = lane & 15;
  const int g     = lane >> 4;
  const int wv    = tid >> 5;
  const int bh    = blockIdx.y;                 // 24 (b,head) pairs
  const int qt    = blockIdx.x * 7 + wv;        // 49 q tiles of 16 rows
  const int q0    = qt * 16;
  const _Float16* qb  = qp + (size_t)bh * QN_ * HD_;
  const _Float16* kb  = kf + (size_t)bh * HW_ * HD_;
  const _Float16* vtb = vt + (size_t)bh * HD_ * HW_;
  const unsigned koff = (unsigned)(uintptr_t)(void*)&Kbuf[0][0];
  const unsigned voff = (unsigned)(uintptr_t)(void*)&Vbuf[0][0];

  // 512 16B chunks per stage (256 K + 256 V) spread over 224 threads
  auto issue_chunk = [&](int c, int s, int j) {
    if (c < 256) {                       // K: chunk -> (key row, 8-half piece)
      const int row = c >> 3, o8 = c & 7;
      async_b128(koff + (unsigned)(s * 4096 + c * 16),
                 kb + (size_t)(j + row) * HD_ + o8 * 8);
    } else {                             // V: chunk -> (dim row, 8-half piece)
      const int c2 = c - 256, d = c2 >> 2, p8 = c2 & 3;
      async_b128(voff + (unsigned)(s * 4096 + c2 * 16),
                 vtb + (size_t)d * HW_ + j + p8 * 8);
    }
  };
  auto issue_stage = [&](int s, int j) {
    issue_chunk(tid, s, j);
    issue_chunk(tid + 224, s, j);
    if (tid < 64) issue_chunk(tid + 448, s, j);
  };

  // q held resident as two A fragments (K = 0..31, 32..63)
  v16h qa[2];
  #pragma unroll
  for (int c = 0; c < 2; ++c) {
    const _Float16* ap = qb + (size_t)(q0 + col16) * HD_ + c * 32 + g * 16;
    #pragma unroll
    for (int e = 0; e < 8; ++e) { qa[c][e] = ap[e]; qa[c][8 + e] = ap[8 + e]; }
  }

  float mrow[8], lrow[8];
  v8f o[4] = {};
  #pragma unroll
  for (int r = 0; r < 8; ++r) { mrow[r] = -1e30f; lrow[r] = 0.f; }

  _Float16* pw = pld[wv];

  issue_stage(0, 0);
  for (int j = 0; j < HW_; j += 32) {
    const int s2 = (j >> 5) & 1;
    if (j + 32 < HW_) {
      issue_stage(s2 ^ 1, j + 32);
      // >=2 ops just issued, in-order completion => all older (stage s2) done
      asm volatile("s_wait_asynccnt 0x2" ::: "memory");
    } else {
      asm volatile("s_wait_asynccnt 0x0" ::: "memory");
    }
    __syncthreads();                       // stage s2 visible to all waves

    const _Float16* Ks = &Kbuf[s2][0];
    const _Float16* Vs = &Vbuf[s2][0];

    v8f s[2];
    #pragma unroll
    for (int tt = 0; tt < 2; ++tt) {
      const _Float16* kp = Ks + (tt * 16 + col16) * 64 + g * 16;
      v16h b0 = *(const v16h*)kp;
      v16h b1 = *(const v16h*)(kp + 32);
      v8f z = {};
      z = wmma16(qa[0], b0, z);
      s[tt] = wmma16(qa[1], b1, z);
    }
    // online softmax: each lane owns rows r + 8*g, reduce over 16 lanes/group
    #pragma unroll
    for (int r = 0; r < 8; ++r) {
      const float mx = redmax16(fmaxf(s[0][r], s[1][r]));
      const float mn = fmaxf(mrow[r], mx);
      const float alpha = __expf(mrow[r] - mn);
      const float p0 = __expf(s[0][r] - mn);
      const float p1 = __expf(s[1][r] - mn);
      s[0][r] = p0; s[1][r] = p1;
      lrow[r] = lrow[r] * alpha + redsum16(p0 + p1);
      mrow[r] = mn;
      #pragma unroll
      for (int f = 0; f < 4; ++f) o[f][r] *= alpha;
    }
    // stage P (C layout) -> wave-private LDS -> reload in A-fragment layout
    #pragma unroll
    for (int r = 0; r < 8; ++r) {
      pw[(r + 8 * g) * 32 + col16]      = (_Float16)s[0][r];
      pw[(r + 8 * g) * 32 + 16 + col16] = (_Float16)s[1][r];
    }
    v16h pa;
    {
      const _Float16* pr = pw + col16 * 32 + g * 16;
      #pragma unroll
      for (int e = 0; e < 8; ++e) { pa[e] = pr[e]; pa[8 + e] = pr[8 + e]; }
    }
    // O += P @ V : 4 x 16x16x32 WMMA covering hd=64, operands from LDS
    #pragma unroll
    for (int f = 0; f < 4; ++f) {
      v16h vfr = *(const v16h*)(Vs + (f * 16 + col16) * 32 + g * 16);
      o[f] = wmma16(pa, vfr, o[f]);
    }
    __syncthreads();                       // all reads of s2 done before reuse
  }

  const int bidx = bh / NH_, head = bh % NH_;
  #pragma unroll
  for (int r = 0; r < 8; ++r) {
    const float inv = 1.f / lrow[r];
    const int pos = q0 + r + 8 * g;
    #pragma unroll
    for (int f = 0; f < 4; ++f)
      attn_out[((size_t)bidx * QN_ + pos) * DOUT_ + head * HD_ + f * 16 + col16] =
          (_Float16)(o[f][r] * inv);
  }
}

// ---------------------------------------------------------------------------
// Kernel 4: out = attn @ W_proj + b_proj.  Grid (nt=12, mtgrp=98), 4 waves
// per block, K=192 -> 6 WMMA per 16x16 tile.
// ---------------------------------------------------------------------------
__global__ __launch_bounds__(128) void proj_gemm(
    const _Float16* __restrict__ attn_in, const float* __restrict__ Wp,
    const float* __restrict__ bp, float* __restrict__ out)
{
  const int lane  = threadIdx.x & 31;
  const int col16 = lane & 15;
  const int g     = lane >> 4;
  const int nt    = blockIdx.x;                          // 0..11
  const int mt    = blockIdx.y * 4 + (threadIdx.x >> 5); // 0..391
  const int ncol  = nt * 16 + col16;

  v8f acc = {};
  #pragma unroll
  for (int kc = 0; kc < 6; ++kc) {
    const int kb = kc * 32;
    v16h a, bb;
    const _Float16* ap = attn_in + (size_t)(mt * 16 + col16) * DOUT_ + kb + g * 8;
    #pragma unroll
    for (int e = 0; e < 8; ++e) { a[e] = ap[e]; a[8 + e] = ap[16 + e]; }
    #pragma unroll
    for (int e = 0; e < 16; ++e)
      bb[e] = (_Float16)Wp[(size_t)(kb + g * 8 + e + (e & 8)) * DOUT_ + ncol];
    acc = wmma16(a, bb, acc);
  }
  const float bias = bp[ncol];
  #pragma unroll
  for (int r = 0; r < 8; ++r) {
    const int m = mt * 16 + r + 8 * g;
    out[(size_t)m * DOUT_ + ncol] = acc[r] + bias;
  }
}

// ---------------------------------------------------------------------------
extern "C" void kernel_launch(void* const* d_in, const int* in_sizes, int n_in,
                              void* d_out, int out_size, void* d_ws, size_t ws_size,
                              hipStream_t stream) {
  (void)in_sizes; (void)n_in; (void)out_size; (void)ws_size;
  const float* x  = (const float*)d_in[0];
  const float* Wq = (const float*)d_in[1];
  const float* bq = (const float*)d_in[2];
  const float* Wp = (const float*)d_in[3];
  const float* bp = (const float*)d_in[4];
  float* out = (float*)d_out;

  char* ws = (char*)d_ws;
  const size_t n_bhld = (size_t)24 * HW_ * HD_;          // 4,816,896 halves
  _Float16* qfull = (_Float16*)(ws);
  _Float16* kf    = (_Float16*)(ws + n_bhld * 2);
  _Float16* vt    = (_Float16*)(ws + n_bhld * 4);        // transposed V
  _Float16* qp    = (_Float16*)(ws + n_bhld * 6);
  _Float16* attn  = (_Float16*)(ws + n_bhld * 6 + (size_t)24 * QN_ * HD_ * 2);
  // total workspace use: ~33.7 MB

  qkv_gemm <<<dim3(36, 49, 8), 128, 0, stream>>>(x, Wq, bq, qfull, kf, vt);
  pool_q   <<<4704, 256, 0, stream>>>(qfull, qp);        // 24*784*64 threads
  flash_attn<<<dim3(7, 24), 224, 0, stream>>>(qp, kf, vt, attn);
  proj_gemm<<<dim3(12, 98), 128, 0, stream>>>(attn, Wp, bp, out);
}
